// DART_53455162966476
// MI455X (gfx1250) — matrix operand order repaired and checked by reading, not compiled, not run on hardware
//
#include <hip/hip_runtime.h>
#include <hip/hip_bf16.h>
#include <math.h>

typedef __bf16 bf16_t;
typedef __attribute__((ext_vector_type(16))) __bf16 v16bf;
typedef __attribute__((ext_vector_type(8)))  __bf16 v8bf;
typedef __attribute__((ext_vector_type(8)))  float  v8f;

#define IDIM  128
#define HDIM  2048
#define KTH   32            // 2*A*A
#define NTH   (IDIM * KTH)  // 4096
#define BATCH 8192

static __device__ __forceinline__ bf16_t f2bf(float f) {
  union { float f; unsigned u; } v; v.f = f;
  unsigned r = v.u + 0x7FFFu + ((v.u >> 16) & 1u);   // round-to-nearest-even
  unsigned short h = (unsigned short)(r >> 16);
  bf16_t o;
  __builtin_memcpy(&o, &h, sizeof(o));
  return o;
}

// ---------------- elementwise conversion ----------------
__global__ void k_f32_to_bf16(const float* __restrict__ in, bf16_t* __restrict__ out, int n) {
  int i = blockIdx.x * blockDim.x + threadIdx.x;
  if (i < n) out[i] = f2bf(in[i]);
}

// mode 0: W0 (2048x128):  keep = (n%127) >= k
// mode 1: W1/W2 (2048^2): keep = (n%127) >= (k%127)
// mode 3: W3 (4096x2048): keep = ((n>>5)-1) >= (k%127)
__global__ void k_mask_bf16(const float* __restrict__ W, bf16_t* __restrict__ out,
                            int N, int Kd, int mode) {
  int i = blockIdx.x * blockDim.x + threadIdx.x;
  if (i >= N * Kd) return;
  int n = i / Kd, k = i - n * Kd;
  bool keep;
  if (mode == 0)      keep = (n % 127) >= k;
  else if (mode == 1) keep = (n % 127) >= (k % 127);
  else                keep = ((n >> 5) - 1) >= (k % 127);
  out[i] = keep ? f2bf(W[i]) : f2bf(0.0f);
}

// ---------------- WMMA GEMM:  C = A(MxK) * W(NxK)^T + bias ----------------
#define BM 128
#define BN 128
#define BK 64
#define LPITCH (BK + 8)   // 72 bf16 = 144B row pitch -> conflict-free frag loads

template<int RELU_BF16_OUT>
__global__ __launch_bounds__(256)
void k_gemm_bf16_wmma(const bf16_t* __restrict__ A,    // (M, Kd) row-major
                      const bf16_t* __restrict__ Bw,   // (N, Kd) row-major
                      const float* __restrict__ bias,  // (N)
                      bf16_t* __restrict__ outb,       // (M, N) bf16 if RELU path
                      float*  __restrict__ outf,       // (M, N) f32 otherwise
                      int M, int N, int Kd) {
  __shared__ bf16_t lsA[2][BM * LPITCH];   // double-buffered
  __shared__ bf16_t lsB[2][BN * LPITCH];

  const int tid   = threadIdx.x;
  const int lane  = tid & 31;
  const int wave  = tid >> 5;
  const int waveM = wave >> 1;   // 0..3  (32 rows each)
  const int waveN = wave & 1;    // 0..1  (64 cols each)
  const int l16   = lane & 15;
  const int lhi   = lane >> 4;   // 0 or 1

  const int mBase = blockIdx.y * BM;
  const int nBase = blockIdx.x * BN;

  v8f acc[2][4];
  const v8f vzero = {0.f, 0.f, 0.f, 0.f, 0.f, 0.f, 0.f, 0.f};
#pragma unroll
  for (int i = 0; i < 2; ++i)
#pragma unroll
    for (int j = 0; j < 4; ++j) acc[i][j] = vzero;

  // staging registers: 4 chunks of 8 bf16 per matrix per thread
  v8bf stA[4], stB[4];
  const int rIdx  = tid >> 3;            // row within 128-row tile (32 rows apart per t)
  const int c8    = (tid & 7) << 3;      // 8-elem column chunk

  auto load_stage = [&](int k0) {
#pragma unroll
    for (int t = 0; t < 4; ++t) {
      int r = rIdx + t * 32;
      stA[t] = *(const v8bf*)(&A [(size_t)(mBase + r) * Kd + k0 + c8]);
      stB[t] = *(const v8bf*)(&Bw[(size_t)(nBase + r) * Kd + k0 + c8]);
    }
  };
  auto store_stage = [&](int buf) {
#pragma unroll
    for (int t = 0; t < 4; ++t) {
      int r = rIdx + t * 32;
      *(v8bf*)(&lsA[buf][r * LPITCH + c8]) = stA[t];
      *(v8bf*)(&lsB[buf][r * LPITCH + c8]) = stB[t];
    }
  };
  auto prefetch_stage = [&](int k0) {   // pull stage k0 into cache (global_prefetch_b8)
    if (k0 < Kd) {
      __builtin_prefetch(&A [(size_t)(mBase + rIdx) * Kd + k0 + c8], 0, 0);
      __builtin_prefetch(&Bw[(size_t)(nBase + rIdx) * Kd + k0 + c8], 0, 0);
    }
  };

  load_stage(0);
  store_stage(0);
  prefetch_stage(BK);
  __syncthreads();

  int cur = 0;
  for (int k0 = 0; k0 < Kd; k0 += BK) {
    const bool has_next = (k0 + BK) < Kd;
    if (has_next) {
      load_stage(k0 + BK);            // global loads overlap with WMMA below
      prefetch_stage(k0 + 2 * BK);
    }

#pragma unroll
    for (int kk = 0; kk < BK; kk += 32) {
      v16bf afrag[2], bfrag[4];
      // A frag (ISA 16-bit A 16x32 layout): row m=l16, K = kbase..+7 and 16+kbase..+7
#pragma unroll
      for (int fm = 0; fm < 2; ++fm) {
        const bf16_t* p = &lsA[cur][(waveM * 32 + fm * 16 + l16) * LPITCH + kk + lhi * 8];
        v8bf lo = *(const v8bf*)(p);
        v8bf hi = *(const v8bf*)(p + 16);
        afrag[fm] = __builtin_shufflevector(lo, hi, 0,1,2,3,4,5,6,7,8,9,10,11,12,13,14,15);
      }
      // B frag: col n=l16, K = kbase..+7 and 16+kbase..+7 (same as A)
#pragma unroll
      for (int fn = 0; fn < 4; ++fn) {
        const bf16_t* p = &lsB[cur][(waveN * 64 + fn * 16 + l16) * LPITCH + kk + lhi * 8];
        v8bf lo = *(const v8bf*)(p);
        v8bf hi = *(const v8bf*)(p + 16);
        bfrag[fn] = __builtin_shufflevector(lo, hi, 0,1,2,3,4,5,6,7,8,9,10,11,12,13,14,15);
      }
#pragma unroll
      for (int fm = 0; fm < 2; ++fm)
#pragma unroll
        for (int fn = 0; fn < 4; ++fn)
          acc[fm][fn] = __builtin_amdgcn_wmma_f32_16x16x32_bf16(
              false, afrag[fm], false, bfrag[fn], (short)0, acc[fm][fn], false, false);
    }

    if (has_next) {
      store_stage(cur ^ 1);           // write the other buffer: no read conflict
      __syncthreads();                // one barrier per stage
      cur ^= 1;
    }
  }

  // epilogue: C/D layout -> element e: m = e + 8*lhi, n = l16
#pragma unroll
  for (int fm = 0; fm < 2; ++fm) {
#pragma unroll
    for (int fn = 0; fn < 4; ++fn) {
      int n = nBase + waveN * 64 + fn * 16 + l16;
      float bv = bias[n];
#pragma unroll
      for (int e = 0; e < 8; ++e) {
        int m = mBase + waveM * 32 + fm * 16 + e + lhi * 8;
        float v = acc[fm][fn][e] + bv;
        if (RELU_BF16_OUT) {
          v = v > 0.f ? v : 0.f;
          outb[(size_t)m * N + n] = f2bf(v);
        } else {
          outf[(size_t)m * N + n] = v;
        }
      }
    }
  }
}

// ---------------- log-space 4x4 chain ----------------
__global__ void k_chain(const float* __restrict__ theta, const float* __restrict__ x,
                        float* __restrict__ out) {
  int b = blockIdx.x * blockDim.x + threadIdx.x;
  if (b >= BATCH) return;
  const float* tb = theta + (size_t)b * NTH;
  const float* xb = x + (size_t)b * IDIM;
  const float CST = 0.91893853320467274178f + 1.38629436111989061883f; // 0.5*ln(2pi)+ln(4)

  float carry[16], lp[16];

  auto compute_lp = [&](int i, float* o) {
    float xv = xb[i];
    const float* mu = tb + i * 32;
    const float* al = mu + 16;
#pragma unroll
    for (int e = 0; e < 16; ++e) {
      float a = al[e];
      float d = (xv - mu[e]) * __expf(-a);
      o[e] = -0.5f * d * d - a - CST;
    }
  };

  compute_lp(1, carry);
  for (int i = 2; i <= 126; ++i) {
    compute_lp(i, lp);
    float nc[16];
#pragma unroll
    for (int r = 0; r < 4; ++r) {
#pragma unroll
      for (int c = 0; c < 4; ++c) {
        float s0 = carry[r * 4 + 0] + lp[0 * 4 + c];
        float s1 = carry[r * 4 + 1] + lp[1 * 4 + c];
        float s2 = carry[r * 4 + 2] + lp[2 * 4 + c];
        float s3 = carry[r * 4 + 3] + lp[3 * 4 + c];
        float mx = fmaxf(fmaxf(s0, s1), fmaxf(s2, s3));
        float sum = __expf(s0 - mx) + __expf(s1 - mx) + __expf(s2 - mx) + __expf(s3 - mx);
        nc[r * 4 + c] = mx + __logf(sum);
      }
    }
#pragma unroll
    for (int e = 0; e < 16; ++e) carry[e] = nc[e];
  }

  compute_lp(127, lp);                       // last column: lp[k*4+0]
  float tv[4];
#pragma unroll
  for (int r = 0; r < 4; ++r) {
    float s0 = carry[r * 4 + 0] + lp[0];
    float s1 = carry[r * 4 + 1] + lp[4];
    float s2 = carry[r * 4 + 2] + lp[8];
    float s3 = carry[r * 4 + 3] + lp[12];
    float mx = fmaxf(fmaxf(s0, s1), fmaxf(s2, s3));
    tv[r] = mx + __logf(__expf(s0 - mx) + __expf(s1 - mx) + __expf(s2 - mx) + __expf(s3 - mx));
  }
  compute_lp(0, lp);                         // first row: lp[0*4+j]
  float s0 = lp[0] + tv[0], s1 = lp[1] + tv[1], s2 = lp[2] + tv[2], s3 = lp[3] + tv[3];
  float mx = fmaxf(fmaxf(s0, s1), fmaxf(s2, s3));
  out[b] = mx + __logf(__expf(s0 - mx) + __expf(s1 - mx) + __expf(s2 - mx) + __expf(s3 - mx));
}

extern "C" void kernel_launch(void* const* d_in, const int* in_sizes, int n_in,
                              void* d_out, int out_size, void* d_ws, size_t ws_size,
                              hipStream_t stream) {
  const float* x  = (const float*)d_in[0];
  const float* W0 = (const float*)d_in[1];
  const float* b0 = (const float*)d_in[2];
  const float* W1 = (const float*)d_in[3];
  const float* b1 = (const float*)d_in[4];
  const float* W2 = (const float*)d_in[5];
  const float* b2 = (const float*)d_in[6];
  const float* W3 = (const float*)d_in[7];
  const float* b3 = (const float*)d_in[8];
  float* out = (float*)d_out;

  char* p = (char*)d_ws;
  auto take = [&](size_t bytes) {
    char* q = p;
    p += (bytes + 255) & ~(size_t)255;
    return q;
  };
  bf16_t* Xb  = (bf16_t*)take((size_t)BATCH * IDIM * 2);
  bf16_t* Wb0 = (bf16_t*)take((size_t)HDIM * IDIM * 2);
  bf16_t* Wb1 = (bf16_t*)take((size_t)HDIM * HDIM * 2);
  bf16_t* Wb2 = (bf16_t*)take((size_t)HDIM * HDIM * 2);
  bf16_t* Wb3 = (bf16_t*)take((size_t)NTH * HDIM * 2);
  bf16_t* Ha  = (bf16_t*)take((size_t)BATCH * HDIM * 2);
  bf16_t* Hb  = (bf16_t*)take((size_t)BATCH * HDIM * 2);
  float*  Th  = (float*)take((size_t)BATCH * NTH * 4);

  int n;
  n = BATCH * IDIM; k_f32_to_bf16<<<(n + 255) / 256, 256, 0, stream>>>(x, Xb, n);
  n = HDIM * IDIM;  k_mask_bf16<<<(n + 255) / 256, 256, 0, stream>>>(W0, Wb0, HDIM, IDIM, 0);
  n = HDIM * HDIM;  k_mask_bf16<<<(n + 255) / 256, 256, 0, stream>>>(W1, Wb1, HDIM, HDIM, 1);
  n = HDIM * HDIM;  k_mask_bf16<<<(n + 255) / 256, 256, 0, stream>>>(W2, Wb2, HDIM, HDIM, 1);
  n = NTH * HDIM;   k_mask_bf16<<<(n + 255) / 256, 256, 0, stream>>>(W3, Wb3, NTH, HDIM, 3);

  dim3 blk(256);
  k_gemm_bf16_wmma<1><<<dim3(HDIM / BN, BATCH / BM), blk, 0, stream>>>(
      Xb, Wb0, b0, Ha, nullptr, BATCH, HDIM, IDIM);
  k_gemm_bf16_wmma<1><<<dim3(HDIM / BN, BATCH / BM), blk, 0, stream>>>(
      Ha, Wb1, b1, Hb, nullptr, BATCH, HDIM, HDIM);
  k_gemm_bf16_wmma<1><<<dim3(HDIM / BN, BATCH / BM), blk, 0, stream>>>(
      Hb, Wb2, b2, Ha, nullptr, BATCH, HDIM, HDIM);
  k_gemm_bf16_wmma<0><<<dim3(NTH / BN, BATCH / BM), blk, 0, stream>>>(
      Ha, Wb3, b3, nullptr, Th, BATCH, NTH, HDIM);

  k_chain<<<BATCH / 256, 256, 0, stream>>>(Th, x, out);
}
